// Model_52999896433232
// MI455X (gfx1250) — matrix operand order repaired and checked by reading, not compiled, not run on hardware
//
#include <hip/hip_runtime.h>
#include <math.h>

// ---------------------------------------------------------------------------
// Model dims (compile-time constants from the reference)
// ---------------------------------------------------------------------------
static constexpr int BB  = 8;
static constexpr int TT  = 12;
static constexpr int NNODE = 512;
static constexpr int HH  = 4;     // heads, hd = 32
static constexpr int LLAY = 3;
static constexpr int PP  = 12;
static constexpr long MTOK = (long)BB * TT * NNODE;   // 49152 tokens

typedef __attribute__((ext_vector_type(16))) __bf16 v16bf;
typedef __attribute__((ext_vector_type(8)))  float  v8f;

// Native bf16 conversion (hardware cvt path) + pair packer for LDS staging.
__device__ __forceinline__ unsigned int pkbf(float x, float y) {
    union { __bf16 h[2]; unsigned int u; } r;
    r.h[0] = (__bf16)x;
    r.h[1] = (__bf16)y;
    return r.u;
}

// ---------------------------------------------------------------------------
// Fragment loaders from LDS (16-bit operands; this code's contiguous K order, not the ISA table order)
// A 16x32 (MxK): lane%16 = row; lanes0-15 hold K=[0,16), lanes16-31 K=[16,32)
//                (same contiguous K order as B)
// B 32x16 (KxN): lane = col; lanes0-15 hold K=[0,16), lanes16-31 K=[16,32)
// ---------------------------------------------------------------------------
__device__ __forceinline__ v16bf frag_A(const unsigned short* rowbase, int hf) {
    union { v16bf v; uint4 q[2]; } r;
    const char* p = (const char*)rowbase;
    r.q[0] = *(const uint4*)(p + hf * 32);
    r.q[1] = *(const uint4*)(p + hf * 32 + 16);
    return r.v;
}
__device__ __forceinline__ v16bf frag_B(const unsigned short* rowbase, int hf) {
    union { v16bf v; uint4 q[2]; } r;
    const char* p = (const char*)rowbase + hf * 32;
    r.q[0] = *(const uint4*)(p);
    r.q[1] = *(const uint4*)(p + 16);
    return r.v;
}

// ---------------------------------------------------------------------------
// GEMM: C[M,N] = epilogue(A[M,K] @ W[N,K]^T + bias[N] (+ Add[M,N]))
// Requires M%128==0, N%128==0, K%32==0 (true for all call sites).
// Block tile 128x128x32, 256 threads = 8 waves (4x2), wave tile 32x64.
// ---------------------------------------------------------------------------
enum { EP_BIAS = 0, EP_GELU = 1, EP_ADD = 2 };
static constexpr int GLDS = 40;   // LDS row stride in halfwords (32 + 8 pad)

template <int EP>
__global__ __launch_bounds__(256)
void gemm_bf16_wmma(const float* __restrict__ A, const float* __restrict__ W,
                    const float* __restrict__ bias, const float* __restrict__ Add,
                    float* __restrict__ C, int M, int N, int K, int ldc, int col_off)
{
    __shared__ __align__(16) unsigned short As[128 * GLDS];
    __shared__ __align__(16) unsigned short Bs[128 * GLDS];

    const int tid  = threadIdx.x;
    const int lane = tid & 31;
    const int wave = tid >> 5;
    const int wr   = wave >> 1;          // 0..3 -> 32-row strip
    const int wc   = wave & 1;           // 0..1 -> 64-col strip
    const int lr   = lane & 15;
    const int hf   = lane >> 4;
    const long blockM = (long)blockIdx.y * 128;
    const long blockN = (long)blockIdx.x * 128;

    v8f acc[2][4];
#pragma unroll
    for (int i = 0; i < 2; i++)
#pragma unroll
        for (int j = 0; j < 4; j++)
#pragma unroll
            for (int e = 0; e < 8; e++) acc[i][j][e] = 0.f;

    const int lrow = tid >> 3;           // 0..31
    const int lcol = (tid & 7) * 4;      // 0..28

    for (int k0 = 0; k0 < K; k0 += 32) {
        // Prefetch next K-tile while staging this one (global_prefetch_b8).
        if (k0 + 32 < K) {
            __builtin_prefetch(A + (blockM + lrow) * (long)K + k0 + 32 + lcol, 0, 1);
            __builtin_prefetch(W + (blockN + lrow) * (long)K + k0 + 32 + lcol, 0, 1);
        }
#pragma unroll
        for (int rr = 0; rr < 128; rr += 32) {
            float4 fa = *(const float4*)(A + (blockM + lrow + rr) * (long)K + k0 + lcol);
            *(uint2*)&As[(lrow + rr) * GLDS + lcol] =
                make_uint2(pkbf(fa.x, fa.y), pkbf(fa.z, fa.w));
            float4 fb = *(const float4*)(W + (blockN + lrow + rr) * (long)K + k0 + lcol);
            *(uint2*)&Bs[(lrow + rr) * GLDS + lcol] =
                make_uint2(pkbf(fb.x, fb.y), pkbf(fb.z, fb.w));
        }
        __syncthreads();

        v16bf af[2], bfr[4];
#pragma unroll
        for (int i = 0; i < 2; i++)
            af[i] = frag_A(&As[(wr * 32 + i * 16 + lr) * GLDS], hf);
#pragma unroll
        for (int j = 0; j < 4; j++)
            bfr[j] = frag_B(&Bs[(wc * 64 + j * 16 + lr) * GLDS], hf);

#pragma unroll
        for (int i = 0; i < 2; i++)
#pragma unroll
            for (int j = 0; j < 4; j++)
                acc[i][j] = __builtin_amdgcn_wmma_f32_16x16x32_bf16(
                    false, af[i], false, bfr[j], (short)0, acc[i][j], false, false);
        __syncthreads();
    }

    // Epilogue. C/D layout: lane = col (lane&15), VGPR r -> row r (+8 for lanes>=16)
#pragma unroll
    for (int i = 0; i < 2; i++) {
#pragma unroll
        for (int j = 0; j < 4; j++) {
            const long gcol = blockN + wc * 64 + j * 16 + lr;
            const float bv  = bias[gcol];
            const long row0 = blockM + wr * 32 + i * 16 + hf * 8;
#pragma unroll
            for (int r = 0; r < 8; r++) {
                float v = acc[i][j][r] + bv;
                if (EP == EP_GELU) v = 0.5f * v * (1.0f + erff(v * 0.70710678118654752f));
                if (EP == EP_ADD)  v += Add[(row0 + r) * (long)N + gcol];
                C[(row0 + r) * (long)ldc + col_off + gcol] = v;
            }
        }
    }
}

// ---------------------------------------------------------------------------
// Spatial attention over nodes (flash-style, one wave per 16-query tile).
// qkv: [M, 384] (q|k|v, each D=128, head h at col h*32). out: [M, 128].
//
// Softmax structure: WMMA scores land in C-layout (lane = kv col, VGPR = row).
// We dump raw scores to an LDS tile, then lane lr owns q-row lr: it reads the
// full 32-score row (8x ds_load_b128), does max/exp/sum in-lane, and builds
// the P fragment for the P@V WMMA directly in registers (A-layout has
// lane = q-row, so no transpose traffic). Only the per-row rescale factors
// for the C-layout O accumulator go through a 16-float LDS broadcast.
// ---------------------------------------------------------------------------
__global__ __launch_bounds__(128)
void spatial_attn(const float* __restrict__ qkv, float* __restrict__ out)
{
    __shared__ __align__(16) float stile[4][16 * 32];  // raw scores, per wave
    __shared__ __align__(16) float sbc[4][16];         // per-row broadcast
    const int tid  = threadIdx.x;
    const int lane = tid & 31;
    const int wave = tid >> 5;
    const int lr   = lane & 15;
    const int hf   = lane >> 4;
    const int qtile = blockIdx.x * 4 + wave;   // 0..31
    const long bt   = blockIdx.y;              // 0..95
    const int  h    = blockIdx.z;              // 0..3
    const float scale = 0.17677669529663687f;  // 1/sqrt(32)

    const float* base = qkv + bt * (long)NNODE * 384 + h * 32;
    const float* kb   = base + 128;
    const float* vb   = base + 256;
    const int klo = hf * 8;

    // Q fragment (A layout), scale folded into bf16 conversion
    v16bf qa;
    {
        const float* qr = base + (long)(qtile * 16 + lr) * 384;
#pragma unroll
        for (int e = 0; e < 8; e++) qa[e]     = (__bf16)(qr[hf * 16 + e] * scale);
#pragma unroll
        for (int e = 0; e < 8; e++) qa[8 + e] = (__bf16)(qr[hf * 16 + 8 + e] * scale);
    }

    float Mrow = -3.0e38f, Lrow = 0.f;         // state for q-row lr (dup per hf)
    v8f o0, o1, zc;
#pragma unroll
    for (int e = 0; e < 8; e++) { o0[e] = 0.f; o1[e] = 0.f; zc[e] = 0.f; }

    float* srow = &stile[wave][lr * 32];

    for (int kv0 = 0; kv0 < NNODE; kv0 += 32) {
        v8f s[2];
#pragma unroll
        for (int t = 0; t < 2; t++) {
            v16bf ka;                           // B layout: lane = kv col
            const float* kr = kb + (long)(kv0 + t * 16 + lr) * 384 + hf * 16;
#pragma unroll
            for (int e = 0; e < 16; e++) ka[e] = (__bf16)kr[e];
            s[t] = __builtin_amdgcn_wmma_f32_16x16x32_bf16(
                false, qa, false, ka, (short)0, zc, false, false);
        }
        // scatter scores to LDS (C layout -> row-major tile)
#pragma unroll
        for (int r = 0; r < 8; r++) {
            const int row = hf * 8 + r;
            stile[wave][row * 32 + lr]      = s[0][r];
            stile[wave][row * 32 + 16 + lr] = s[1][r];
        }
        // lane lr processes its whole row
        float sv[32];
#pragma unroll
        for (int c4 = 0; c4 < 8; c4++) {
            const float4 t4 = *(const float4*)(srow + c4 * 4);
            sv[c4 * 4 + 0] = t4.x; sv[c4 * 4 + 1] = t4.y;
            sv[c4 * 4 + 2] = t4.z; sv[c4 * 4 + 3] = t4.w;
        }
        float mx = sv[0];
#pragma unroll
        for (int c = 1; c < 32; c++) mx = fmaxf(mx, sv[c]);
        const float mnew = fmaxf(Mrow, mx);
        const float sf   = __expf(Mrow - mnew);
        float ssum = 0.f;
        float pv[32];
#pragma unroll
        for (int c = 0; c < 32; c++) { pv[c] = __expf(sv[c] - mnew); ssum += pv[c]; }
        Lrow = Lrow * sf + ssum;
        Mrow = mnew;

        // broadcast per-row rescale factor (both halves write same value)
        sbc[wave][lr] = sf;

        // build P fragment directly in A layout (lane = row lr)
        v16bf pf;
#pragma unroll
        for (int e = 0; e < 8; e++) pf[e]     = (__bf16)pv[hf * 16 + e];
#pragma unroll
        for (int e = 0; e < 8; e++) pf[8 + e] = (__bf16)pv[hf * 16 + 8 + e];

        // rescale O accumulators (C layout: VGPR r -> row hf*8+r)
        const float4 sf0 = *(const float4*)&sbc[wave][hf * 8];
        const float4 sf1 = *(const float4*)&sbc[wave][hf * 8 + 4];
        o0[0] *= sf0.x; o0[1] *= sf0.y; o0[2] *= sf0.z; o0[3] *= sf0.w;
        o0[4] *= sf1.x; o0[5] *= sf1.y; o0[6] *= sf1.z; o0[7] *= sf1.w;
        o1[0] *= sf0.x; o1[1] *= sf0.y; o1[2] *= sf0.z; o1[3] *= sf0.w;
        o1[4] *= sf1.x; o1[5] *= sf1.y; o1[6] *= sf1.z; o1[7] *= sf1.w;

#pragma unroll
        for (int t = 0; t < 2; t++) {
            v16bf va;                           // B layout: lane = e col
#pragma unroll
            for (int j = 0; j < 16; j++)
                va[j] = (__bf16)vb[(long)(kv0 + hf * 16 + j) * 384 + t * 16 + lr];
            v8f& ot = t ? o1 : o0;
            ot = __builtin_amdgcn_wmma_f32_16x16x32_bf16(
                false, pf, false, va, (short)0, ot, false, false);
        }
    }

    // final normalization: broadcast 1/Lrow per row, then scale + store
    sbc[wave][lr] = 1.0f / Lrow;
    const float4 i0 = *(const float4*)&sbc[wave][hf * 8];
    const float4 i1 = *(const float4*)&sbc[wave][hf * 8 + 4];
    const float inv[8] = { i0.x, i0.y, i0.z, i0.w, i1.x, i1.y, i1.z, i1.w };
#pragma unroll
    for (int r = 0; r < 8; r++) {
        const long row = bt * NNODE + qtile * 16 + hf * 8 + r;
        out[row * 128 + h * 32 + lr]      = o0[r] * inv[r];
        out[row * 128 + h * 32 + 16 + lr] = o1[r] * inv[r];
    }
}

// ---------------------------------------------------------------------------
// Temporal attention over T=12 (negligible FLOPs -> scalar VALU).
// ---------------------------------------------------------------------------
__global__ __launch_bounds__(256)
void temporal_attn(const float* __restrict__ qkv, float* __restrict__ out)
{
    const long idx = (long)blockIdx.x * blockDim.x + threadIdx.x;  // B*N*H*T
    const int tq = (int)(idx % TT);
    const int h  = (int)((idx / TT) % HH);
    const long n = (idx / (TT * HH)) % NNODE;
    const long b = idx / ((long)TT * HH * NNODE);
    const float scale = 0.17677669529663687f;
    const long rq = (b * TT + tq) * NNODE + n;

    float qv[32];
    const float* q = qkv + rq * 384 + h * 32;
#pragma unroll
    for (int e = 0; e < 32; e++) qv[e] = q[e];

    float sc[TT];
    float mx = -3e38f;
    for (int tk = 0; tk < TT; tk++) {
        const float* kr = qkv + ((b * TT + tk) * NNODE + n) * 384 + 128 + h * 32;
        float d = 0.f;
#pragma unroll
        for (int e = 0; e < 32; e++) d += qv[e] * kr[e];
        sc[tk] = d * scale;
        mx = fmaxf(mx, sc[tk]);
    }
    float sum = 0.f;
    for (int tk = 0; tk < TT; tk++) { sc[tk] = __expf(sc[tk] - mx); sum += sc[tk]; }
    const float inv = 1.0f / sum;

    float accv[32];
#pragma unroll
    for (int e = 0; e < 32; e++) accv[e] = 0.f;
    for (int tk = 0; tk < TT; tk++) {
        const float p = sc[tk] * inv;
        const float* vr = qkv + ((b * TT + tk) * NNODE + n) * 384 + 256 + h * 32;
#pragma unroll
        for (int e = 0; e < 32; e++) accv[e] += p * vr[e];
    }
    float* o = out + rq * 128 + h * 32;
#pragma unroll
    for (int e = 0; e < 32; e++) o[e] = accv[e];
}

// ---------------------------------------------------------------------------
// LayerNorm over D=128: one wave per token, 4 elements/lane.
// ---------------------------------------------------------------------------
__global__ __launch_bounds__(256)
void layernorm_k(const float* __restrict__ x, const float* __restrict__ g,
                 const float* __restrict__ b, float* __restrict__ y)
{
    const int lane = threadIdx.x & 31;
    const int wave = threadIdx.x >> 5;
    const long tok = (long)blockIdx.x * 8 + wave;
    const float4 v = *(const float4*)(x + tok * 128 + lane * 4);
    float s = v.x + v.y + v.z + v.w;
#pragma unroll
    for (int d = 1; d < 32; d <<= 1) s += __shfl_xor(s, d, 32);
    const float mu = s * (1.0f / 128.0f);
    const float cx = v.x - mu, cy = v.y - mu, cz = v.z - mu, cw = v.w - mu;
    float q = cx * cx + cy * cy + cz * cz + cw * cw;
#pragma unroll
    for (int d = 1; d < 32; d <<= 1) q += __shfl_xor(q, d, 32);
    const float rstd = rsqrtf(q * (1.0f / 128.0f) + 1e-5f);
    const float4 gg = *(const float4*)(g + lane * 4);
    const float4 bb = *(const float4*)(b + lane * 4);
    float4 o;
    o.x = cx * rstd * gg.x + bb.x;
    o.y = cy * rstd * gg.y + bb.y;
    o.z = cz * rstd * gg.z + bb.z;
    o.w = cw * rstd * gg.w + bb.w;
    *(float4*)(y + tok * 128 + lane * 4) = o;
}

// ---------------------------------------------------------------------------
// Embedding: h = x @ tokW^T + tokb + sinusoidal PE over T.
// ---------------------------------------------------------------------------
__global__ __launch_bounds__(256)
void embed_kernel(const float* __restrict__ x, const float* __restrict__ tokW,
                  const float* __restrict__ tokb, float* __restrict__ h)
{
    const long idx = (long)blockIdx.x * blockDim.x + threadIdx.x;  // M*128
    const long m = idx >> 7;
    const int  d = (int)(idx & 127);
    const int  t = (int)((m / NNODE) % TT);
    const float* xr = x + m * 3;
    const float* wr = tokW + d * 3;
    float s = tokb[d] + xr[0] * wr[0] + xr[1] * wr[1] + xr[2] * wr[2];
    const float freq = __expf(-(float)(d & ~1) * 0.07195578239529289f); // ln(1e4)/128
    const float ang  = (float)t * freq;
    s += (d & 1) ? __cosf(ang) : __sinf(ang);
    h[idx] = s;
}

// ---------------------------------------------------------------------------
// Head: y[b,p,n] = sum_c relu(sum_t xc[b,t,n,c]*e1W[p,t] + e1b[p]) * e2W[c] + e2b
// One 128-thread block per (b,p,n); xc = concat(out1,out2,out3) over c.
// ---------------------------------------------------------------------------
__global__ __launch_bounds__(128)
void head_kernel(const float* __restrict__ o1, const float* __restrict__ o2,
                 const float* __restrict__ o3, const float* __restrict__ w1,
                 const float* __restrict__ b1, const float* __restrict__ w2,
                 const float* __restrict__ b2, float* __restrict__ out)
{
    __shared__ float red[4];
    const long blk = blockIdx.x;
    const long n = blk % NNODE;
    const int  p = (int)((blk / NNODE) % PP);
    const long b = blk / ((long)NNODE * PP);
    const int tid = threadIdx.x, lane = tid & 31, wave = tid >> 5;

    float partial = 0.f;
#pragma unroll
    for (int li = 0; li < 3; li++) {
        const int c = tid + li * 128;
        const float* src = (li == 0) ? o1 : (li == 1) ? o2 : o3;
        float a = b1[p];
#pragma unroll
        for (int t = 0; t < TT; t++)
            a += src[((b * TT + t) * NNODE + n) * 128 + tid] * w1[p * TT + t];
        a = fmaxf(a, 0.f);
        partial += a * w2[c];
    }
#pragma unroll
    for (int d = 1; d < 32; d <<= 1) partial += __shfl_xor(partial, d, 32);
    if (lane == 0) red[wave] = partial;
    __syncthreads();
    if (tid == 0)
        out[(b * PP + p) * NNODE + n] = red[0] + red[1] + red[2] + red[3] + b2[0];
}

// ---------------------------------------------------------------------------
// Host: launch sequence (graph-capture safe; only kernel launches on stream)
// ---------------------------------------------------------------------------
static inline void launch_gemm(int ep, const float* A, const float* W,
                               const float* bias, const float* Add, float* C,
                               int M, int N, int K, int ldc, int col_off,
                               hipStream_t s)
{
    dim3 grid(N / 128, M / 128);
    if (ep == EP_GELU)
        gemm_bf16_wmma<EP_GELU><<<grid, 256, 0, s>>>(A, W, bias, Add, C, M, N, K, ldc, col_off);
    else if (ep == EP_ADD)
        gemm_bf16_wmma<EP_ADD><<<grid, 256, 0, s>>>(A, W, bias, Add, C, M, N, K, ldc, col_off);
    else
        gemm_bf16_wmma<EP_BIAS><<<grid, 256, 0, s>>>(A, W, bias, Add, C, M, N, K, ldc, col_off);
}

extern "C" void kernel_launch(void* const* d_in, const int* in_sizes, int n_in,
                              void* d_out, int out_size, void* d_ws, size_t ws_size,
                              hipStream_t stream)
{
    (void)in_sizes; (void)n_in; (void)out_size; (void)ws_size;
    const float* x     = (const float*)d_in[0];
    const float* tokW  = (const float*)d_in[1];
    const float* tokb  = (const float*)d_in[2];
    const float* tqkvW = (const float*)d_in[3];
    const float* tqkvb = (const float*)d_in[4];
    const float* tpW   = (const float*)d_in[5];
    const float* tpb   = (const float*)d_in[6];
    const float* gqkvW = (const float*)d_in[7];
    const float* gqkvb = (const float*)d_in[8];
    const float* gpW   = (const float*)d_in[9];
    const float* gpb   = (const float*)d_in[10];
    const float* sqkvW = (const float*)d_in[11];
    const float* sqkvb = (const float*)d_in[12];
    const float* spW   = (const float*)d_in[13];
    const float* spb   = (const float*)d_in[14];
    const float* resW  = (const float*)d_in[15];
    const float* resb  = (const float*)d_in[16];
    const float* normW = (const float*)d_in[17];
    const float* normb = (const float*)d_in[18];
    const float* fc1W  = (const float*)d_in[19];
    const float* fc1b  = (const float*)d_in[20];
    const float* fc2W  = (const float*)d_in[21];
    const float* fc2b  = (const float*)d_in[22];
    const float* e1W   = (const float*)d_in[23];
    const float* e1b   = (const float*)d_in[24];
    const float* e2W   = (const float*)d_in[25];
    const float* e2b   = (const float*)d_in[26];
    float* out = (float*)d_out;

    const long M = MTOK;                       // 49152, multiple of 128
    float* ws   = (float*)d_ws;
    float* outs = ws;                          // 4 slots of [M,128] (emb + 3 layers)
    float* bqkv = outs + 4 * M * 128;          // [M,384]  (also reused as fc1 out [M,256])
    float* ba   = bqkv + M * 384;              // [M,128]  attention raw output / fc2 out
    float* bx   = ba + M * 128;                // [M,128]  temporal-proj output / pre-LN
    float* bc   = bx + M * 128;                // [M,256]  concat(geo|sem)

    const int Mi = (int)M;

    embed_kernel<<<(int)(M * 128 / 256), 256, 0, stream>>>(x, tokW, tokb, outs);

    const dim3 sgrid(NNODE / 64, BB * TT, HH);
    for (int l = 0; l < LLAY; l++) {
        const float* R = outs + (long)l * M * 128;        // residual (layer input)
        float* OUT     = outs + (long)(l + 1) * M * 128;  // layer output slot

        // --- temporal attention ---
        launch_gemm(EP_BIAS, R, tqkvW + (long)l * 3 * 128 * 128, tqkvb + l * 384,
                    nullptr, bqkv, Mi, 384, 128, 384, 0, stream);
        temporal_attn<<<768, 256, 0, stream>>>(bqkv, ba);
        launch_gemm(EP_BIAS, ba, tpW + (long)l * 128 * 128, tpb + l * 128,
                    nullptr, bx, Mi, 128, 128, 128, 0, stream);

        // --- geo spatial attention -> cat[:, 0:128] ---
        launch_gemm(EP_BIAS, bx, gqkvW + (long)l * 3 * 128 * 128, gqkvb + l * 384,
                    nullptr, bqkv, Mi, 384, 128, 384, 0, stream);
        spatial_attn<<<sgrid, 128, 0, stream>>>(bqkv, ba);
        launch_gemm(EP_BIAS, ba, gpW + (long)l * 128 * 128, gpb + l * 128,
                    nullptr, bc, Mi, 128, 128, 256, 0, stream);

        // --- sem spatial attention -> cat[:, 128:256] ---
        launch_gemm(EP_BIAS, bx, sqkvW + (long)l * 3 * 128 * 128, sqkvb + l * 384,
                    nullptr, bqkv, Mi, 384, 128, 384, 0, stream);
        spatial_attn<<<sgrid, 128, 0, stream>>>(bqkv, ba);
        launch_gemm(EP_BIAS, ba, spW + (long)l * 128 * 128, spb + l * 128,
                    nullptr, bc, Mi, 128, 128, 256, 128, stream);

        // --- MLP: fc1 (GELU) -> fc2 ---
        launch_gemm(EP_GELU, bc, fc1W + (long)l * 256 * 256, fc1b + l * 256,
                    nullptr, bqkv, Mi, 256, 256, 256, 0, stream);
        launch_gemm(EP_BIAS, bqkv, fc2W + (long)l * 128 * 256, fc2b + l * 128,
                    nullptr, ba, Mi, 128, 256, 128, 0, stream);

        // --- pre = res_proj(residual) + resb + mlp, then LayerNorm ---
        launch_gemm(EP_ADD, R, resW + (long)l * 128 * 128, resb + l * 128,
                    ba, bx, Mi, 128, 128, 128, 0, stream);
        layernorm_k<<<(int)(M / 8), 256, 0, stream>>>(bx, normW + l * 128,
                                                      normb + l * 128, OUT);
    }

    head_kernel<<<(int)((long)BB * PP * NNODE), 128, 0, stream>>>(
        outs + M * 128, outs + 2 * M * 128, outs + 3 * M * 128,
        e1W, e1b, e2W, e2b, out);
}
